// Head_79843442032792
// MI455X (gfx1250) — matrix operand order repaired and checked by reading, not compiled, not run on hardware
//
#include <hip/hip_runtime.h>
#include <hip/hip_bf16.h>

// ---------------------------------------------------------------------------
// Single-head attention for MI455X (gfx1250), bf16 WMMA + f32 accumulation.
//   q = (x@Wq + bq) * sqrt(C)   (scale folded into q)
//   k = (x@Wk + bk)
//   v = (x@Wv + bv), stored transposed vT[B][E][T]
//   out = softmax(q @ k^T) @ v   (flash-attention, online softmax)
// k/v blocks are staged into LDS by the Tensor Data Mover (double-buffered).
// ---------------------------------------------------------------------------

#define BATCH   4
#define T_SEQ   2048
#define CDIM    1024
#define EDIM    256

typedef __attribute__((ext_vector_type(16))) __bf16 v16bf;
typedef __attribute__((ext_vector_type(8)))  __bf16 v8bf;
typedef __attribute__((ext_vector_type(8)))  float  v8f;
typedef __attribute__((ext_vector_type(4)))  unsigned int v4u;
typedef __attribute__((ext_vector_type(4)))  int v4i;
typedef __attribute__((ext_vector_type(8)))  int v8i;

// ---- LDS staging layout (dynamic shared memory) ----
// k buffer: 64 rows x 256 bf16, TDM-padded +4 dwords per 128 dwords -> stride 264 halfs
// v buffer: 256 rows x 64 bf16, TDM-padded +4 dwords per 32 dwords  -> stride 72 halfs
#define K_STRIDE_H   264            // halfs
#define V_STRIDE_H   72             // halfs
#define KBUF_BYTES   (64 * K_STRIDE_H * 2)    // 33792
#define VBUF_BYTES   (256 * V_STRIDE_H * 2)   // 36864
#define KBUF_OFF     0
#define VBUF_OFF     (2 * KBUF_BYTES)         // 67584
#define PLDS_OFF     (VBUF_OFF + 2 * VBUF_BYTES)   // 141312
#define SMEM_BYTES   (PLDS_OFF + 8 * 16 * 72 * 2)  // 159744

__device__ __forceinline__ v8f wmma_bf16(v16bf a, v16bf b, v8f c) {
    return __builtin_amdgcn_wmma_f32_16x16x32_bf16(false, a, false, b,
                                                   (short)0, c, false, false);
}

// A-fragment (16x32, M x K), contiguous k order (same as load_b/load_b16):
// lane L: m = L%16 ; halves 0-15 -> k = (L/16)*16 + i
__device__ __forceinline__ v16bf load_a(const __bf16* base, int ld, int lane) {
    const __bf16* p = base + (size_t)(lane & 15) * ld + ((lane >> 4) << 4);
    v8bf lo = *(const v8bf*)(p);
    v8bf hi = *(const v8bf*)(p + 8);
    v16bf a;
#pragma unroll
    for (int i = 0; i < 8; ++i) { a[i] = lo[i]; a[i + 8] = hi[i]; }
    return a;
}

// B-fragment (32x16, K x N) from an N-major buffer, one 32B contiguous load.
__device__ __forceinline__ v16bf load_b(const __bf16* baseT, int ld, int lane) {
    const __bf16* p = baseT + (size_t)(lane & 15) * ld + ((lane >> 4) << 4);
    return *(const v16bf*)p;
}

// Same as load_b but split into two 16B loads (LDS rows are only 16B aligned).
__device__ __forceinline__ v16bf load_b16(const __bf16* baseT, int ld, int lane) {
    const __bf16* p = baseT + (size_t)(lane & 15) * ld + ((lane >> 4) << 4);
    v8bf lo = *(const v8bf*)(p);
    v8bf hi = *(const v8bf*)(p + 8);
    v16bf r;
#pragma unroll
    for (int i = 0; i < 8; ++i) { r[i] = lo[i]; r[i + 8] = hi[i]; }
    return r;
}

// ---- Tensor Data Mover: 2D tile load, D# built per CDNA5 ISA ch.8 ----
// pad_int: pad every 2^(pad_int+1) dwords; pad_amt: (pad_amt+1) dwords of padding.
__device__ __forceinline__ void tdm_load_2d(unsigned lds_addr, const void* gptr,
                                            unsigned tile_d0, unsigned tile_d1,
                                            unsigned tens_d0, unsigned tens_d1,
                                            unsigned stride_elems,
                                            unsigned pad_int, unsigned pad_amt) {
    unsigned long long ga = (unsigned long long)gptr;
    v4u g0;
    g0[0] = 1u;                                    // count=1, user descriptor
    g0[1] = lds_addr;                              // LDS byte address
    g0[2] = (unsigned)ga;                          // global_addr[31:0]
    g0[3] = (unsigned)((ga >> 32) & 0x01FFFFFFu) | (2u << 30);  // addr[56:32] | type=2
    v8i g1;
    g1[0] = (int)((1u << 16) |                     // data_size = 1 -> 2 bytes
                  (1u << 20) |                     // pad_enable
                  (pad_int << 22) | (pad_amt << 25));
    g1[1] = (int)((tens_d0 & 0xFFFFu) << 16);      // tensor_dim0[15:0]
    g1[2] = (int)(((tens_d0 >> 16) & 0xFFFFu) | ((tens_d1 & 0xFFFFu) << 16));
    g1[3] = (int)(((tens_d1 >> 16) & 0xFFFFu) | ((tile_d0 & 0xFFFFu) << 16));
    g1[4] = (int)(tile_d1 & 0xFFFFu);              // tile_dim1; tile_dim2 = 0
    g1[5] = (int)stride_elems;                     // tensor_dim0_stride[31:0]
    g1[6] = 0;
    g1[7] = 0;
    v4i z4 = {0, 0, 0, 0};
    v8i z8 = {0, 0, 0, 0, 0, 0, 0, 0};
    __builtin_amdgcn_tensor_load_to_lds(g0, g1, z4, z4, z8, 0);
}

__device__ __forceinline__ unsigned lds_offset_of(const void* p) {
    return (unsigned)(unsigned long long)p;        // low 32 bits = LDS byte offset
}

// ---------------------------------------------------------------------------
// Kernel 0: convert x to bf16; convert + transpose weights to Wt[3][E][C] bf16.
// ---------------------------------------------------------------------------
__global__ void pack_kernel(const float* __restrict__ x,
                            const float* __restrict__ Wq,
                            const float* __restrict__ Wk,
                            const float* __restrict__ Wv,
                            __bf16* __restrict__ xb,
                            __bf16* __restrict__ Wt) {
    const int tid    = blockIdx.x * blockDim.x + threadIdx.x;
    const int stride = gridDim.x * blockDim.x;
    const int NX = BATCH * T_SEQ * CDIM;
    for (int i = tid; i < NX; i += stride) xb[i] = (__bf16)x[i];
    const int NW = 3 * EDIM * CDIM;
    for (int i = tid; i < NW; i += stride) {
        int m = i / (EDIM * CDIM);
        int r = i % (EDIM * CDIM);
        int e = r / CDIM;
        int c = r % CDIM;
        const float* W = (m == 0) ? Wq : (m == 1) ? Wk : Wv;
        Wt[i] = (__bf16)W[c * EDIM + e];   // Wt[m][e][c] = W[c][e]
    }
}

// ---------------------------------------------------------------------------
// Kernel 1: QKV projection. One wave computes a 32x64 output tile of q, k or v.
// Manual 2-slice ping-pong: loads of slice i+1 overlap WMMAs of slice i, and
// the period-2 rotation maps registers onto themselves (no phi copies).
// ---------------------------------------------------------------------------
__global__ __launch_bounds__(256) void proj_kernel(
        const __bf16* __restrict__ xb,   // [B*T][C]
        const __bf16* __restrict__ Wt,   // [3][E][C]
        const float*  __restrict__ bq,
        const float*  __restrict__ bk,
        const float*  __restrict__ bvp,
        __bf16* __restrict__ q,          // [B*T][E]
        __bf16* __restrict__ k,          // [B*T][E]
        __bf16* __restrict__ vT) {       // [B][E][T]
    const int lane = threadIdx.x & 31;
    const int wg   = blockIdx.x * 8 + (threadIdx.x >> 5);
    const int rt   = wg / 12;            // 256 row tiles of 32 over B*T
    const int rem  = wg % 12;
    const int mat  = rem >> 2;           // 0=q 1=k 2=v
    const int col0 = (rem & 3) * 64;
    const int row0 = rt * 32;

    v8f acc[2][4];
#pragma unroll
    for (int h = 0; h < 2; ++h)
#pragma unroll
        for (int t = 0; t < 4; ++t)
            acc[h][t] = v8f{0.f,0.f,0.f,0.f,0.f,0.f,0.f,0.f};

    const __bf16* wbase = Wt + (size_t)mat * EDIM * CDIM;
    const __bf16* xr0   = xb + (size_t)row0 * CDIM;
    const __bf16* xr1   = xr0 + (size_t)16 * CDIM;

    auto load_slice = [&](v16bf& A0, v16bf& A1, v16bf (&B)[4], int cc) {
        A0 = load_a(xr0 + cc, CDIM, lane);
        A1 = load_a(xr1 + cc, CDIM, lane);
#pragma unroll
        for (int t = 0; t < 4; ++t)
            B[t] = load_b(wbase + (size_t)(col0 + t * 16) * CDIM + cc, CDIM, lane);
    };
    auto mma_slice = [&](const v16bf& A0, const v16bf& A1, const v16bf (&B)[4]) {
#pragma unroll
        for (int t = 0; t < 4; ++t) {
            acc[0][t] = wmma_bf16(A0, B[t], acc[0][t]);
            acc[1][t] = wmma_bf16(A1, B[t], acc[1][t]);
        }
    };

    v16bf pA0, pA1, pB[4], qA0, qA1, qB[4];
    load_slice(pA0, pA1, pB, 0);
    for (int c = 0; c < CDIM - 64; c += 64) {
        load_slice(qA0, qA1, qB, c + 32);
        mma_slice(pA0, pA1, pB);
        load_slice(pA0, pA1, pB, c + 64);
        mma_slice(qA0, qA1, qB);
    }
    load_slice(qA0, qA1, qB, CDIM - 32);
    mma_slice(pA0, pA1, pB);
    mma_slice(qA0, qA1, qB);

    const float* bias = (mat == 0) ? bq : (mat == 1) ? bk : bvp;
    const float  sc   = (mat == 0) ? 32.0f : 1.0f;   // sqrt(1024)
    const int m0 = (lane >> 4) * 8;
    const int n  = lane & 15;
#pragma unroll
    for (int h = 0; h < 2; ++h) {
#pragma unroll
        for (int t = 0; t < 4; ++t) {
            const int e = col0 + t * 16 + n;
            const float be = bias[e];
#pragma unroll
            for (int r = 0; r < 8; ++r) {
                const float val = (acc[h][t][r] + be) * sc;
                const int row = row0 + h * 16 + m0 + r;
                if (mat == 0)      q[(size_t)row * EDIM + e] = (__bf16)val;
                else if (mat == 1) k[(size_t)row * EDIM + e] = (__bf16)val;
                else {
                    const int bb = row >> 11;
                    const int tt = row & (T_SEQ - 1);
                    vT[((size_t)(bb * EDIM + e)) * T_SEQ + tt] = (__bf16)val;
                }
            }
        }
    }
}

// ---------------------------------------------------------------------------
// Kernel 2: flash attention. Block = 8 waves; wave owns 16 q rows.
// grid = (T/128, B). k/v consumed in 64-wide blocks, TDM-staged in LDS,
// double-buffered; DMA of block i+1 overlaps compute of block i.
// ---------------------------------------------------------------------------
__global__ __launch_bounds__(256) void attn_kernel(
        const __bf16* __restrict__ qb,   // [B*T][E], pre-scaled by sqrt(C)
        const __bf16* __restrict__ kb,   // [B*T][E]
        const __bf16* __restrict__ vT,   // [B][E][T]
        float* __restrict__ out) {       // [B*T][E] f32
    extern __shared__ char smem[];
    __bf16* kb0 = (__bf16*)(smem + KBUF_OFF);
    __bf16* kb1 = (__bf16*)(smem + KBUF_OFF + KBUF_BYTES);
    __bf16* vb0 = (__bf16*)(smem + VBUF_OFF);
    __bf16* vb1 = (__bf16*)(smem + VBUF_OFF + VBUF_BYTES);
    __bf16* plds_w = (__bf16*)(smem + PLDS_OFF) + (threadIdx.x >> 5) * 16 * 72;

    const int lane = threadIdx.x & 31;
    const int wave = threadIdx.x >> 5;
    const int b    = blockIdx.y;
    const int row0 = blockIdx.x * 128 + wave * 16;

    const __bf16* qbase = qb + ((size_t)(b * T_SEQ + row0)) * EDIM;
    const __bf16* kbase = kb + (size_t)b * T_SEQ * EDIM;
    const __bf16* vbase = vT + (size_t)b * EDIM * T_SEQ;

    const int NBLK = T_SEQ / 64;

    // Kick off DMA of block 0 into buffer 0 (one wave issues; EXEC-independent DMA).
    if (wave == 0) {
        tdm_load_2d(lds_offset_of(kb0), kbase, /*tile*/256, 64,
                    /*tensor*/256, T_SEQ, /*stride*/EDIM, /*pad*/6, 3);
        tdm_load_2d(lds_offset_of(vb0), vbase, /*tile*/64, 256,
                    /*tensor*/T_SEQ, EDIM, /*stride*/T_SEQ, /*pad*/4, 3);
    }

    // Preload all 8 q A-fragments (K = E = 256).
    v16bf qa[8];
#pragma unroll
    for (int kc = 0; kc < 8; ++kc)
        qa[kc] = load_a(qbase + kc * 32, EDIM, lane);

    v8f o[16];
#pragma unroll
    for (int i = 0; i < 16; ++i) o[i] = v8f{0.f,0.f,0.f,0.f,0.f,0.f,0.f,0.f};

    float mst[8], lst[8];
#pragma unroll
    for (int r = 0; r < 8; ++r) { mst[r] = -1e30f; lst[r] = 0.f; }

    for (int kblk = 0; kblk < NBLK; ++kblk) {
        const int cur = kblk & 1;
        if (wave == 0) __builtin_amdgcn_s_wait_tensorcnt(0);
        __syncthreads();   // current buffers ready for everyone; previous reads done
        if (wave == 0 && kblk + 1 < NBLK) {
            tdm_load_2d(lds_offset_of(cur ? kb0 : kb1),
                        kbase + (size_t)(kblk + 1) * 64 * EDIM,
                        256, 64, 256, T_SEQ, EDIM, 6, 3);
            tdm_load_2d(lds_offset_of(cur ? vb0 : vb1),
                        vbase + (size_t)(kblk + 1) * 64,
                        64, 256, T_SEQ, EDIM, T_SEQ, 4, 3);
        }
        const __bf16* kc_lds = cur ? kb1 : kb0;
        const __bf16* vc_lds = cur ? vb1 : vb0;

        // ---- S = q @ k^T (scale folded into q) ----
        v8f s[4];
#pragma unroll
        for (int t = 0; t < 4; ++t) s[t] = v8f{0.f,0.f,0.f,0.f,0.f,0.f,0.f,0.f};
#pragma unroll
        for (int kc = 0; kc < 8; ++kc) {
#pragma unroll
            for (int t = 0; t < 4; ++t) {
                v16bf bf = load_b16(kc_lds + (size_t)(t * 16) * K_STRIDE_H + kc * 32,
                                    K_STRIDE_H, lane);
                s[t] = wmma_bf16(qa[kc], bf, s[t]);
            }
        }

        // ---- online softmax (row m = r + 8*(lane/16), replicated over 16 lanes) ----
        float corr[8];
#pragma unroll
        for (int r = 0; r < 8; ++r) {
            float mx = fmaxf(fmaxf(s[0][r], s[1][r]), fmaxf(s[2][r], s[3][r]));
#pragma unroll
            for (int msk = 1; msk < 16; msk <<= 1)
                mx = fmaxf(mx, __shfl_xor(mx, msk, 32));
            const float mn = fmaxf(mst[r], mx);
            const float c  = __expf(mst[r] - mn);
            mst[r] = mn;
            float rs = 0.f;
#pragma unroll
            for (int t = 0; t < 4; ++t) {
                const float p = __expf(s[t][r] - mn);
                s[t][r] = p;
                rs += p;
            }
#pragma unroll
            for (int msk = 1; msk < 16; msk <<= 1)
                rs += __shfl_xor(rs, msk, 32);
            lst[r] = lst[r] * c + rs;
            corr[r] = c;
        }
#pragma unroll
        for (int et = 0; et < 16; ++et)
#pragma unroll
            for (int r = 0; r < 8; ++r) o[et][r] *= corr[r];

        // ---- P (C-layout f32) -> per-wave LDS bf16 tile (padded stride 72) ----
        const int mrow = (lane >> 4) * 8;
#pragma unroll
        for (int t = 0; t < 4; ++t)
#pragma unroll
            for (int r = 0; r < 8; ++r)
                plds_w[(mrow + r) * 72 + t * 16 + (lane & 15)] = (__bf16)s[t][r];
        // DS ops are in-order within a wave: no barrier needed before re-reading.

        // ---- O += P @ v ----
#pragma unroll
        for (int kc2 = 0; kc2 < 2; ++kc2) {
            v16bf pa = load_a(plds_w + kc2 * 32, 72, lane);
#pragma unroll
            for (int et = 0; et < 16; ++et) {
                v16bf bv = load_b16(vc_lds + (size_t)(et * 16) * V_STRIDE_H + kc2 * 32,
                                    V_STRIDE_H, lane);
                o[et] = wmma_bf16(pa, bv, o[et]);
            }
        }
    }

    // ---- epilogue: divide by row sums, store f32 ----
#pragma unroll
    for (int r = 0; r < 8; ++r) lst[r] = 1.0f / lst[r];
    const int m0 = (lane >> 4) * 8;
#pragma unroll
    for (int et = 0; et < 16; ++et)
#pragma unroll
        for (int r = 0; r < 8; ++r)
            out[((size_t)(b * T_SEQ + row0 + m0 + r)) * EDIM + et * 16 + (lane & 15)] =
                o[et][r] * lst[r];
}

// ---------------------------------------------------------------------------
extern "C" void kernel_launch(void* const* d_in, const int* in_sizes, int n_in,
                              void* d_out, int out_size, void* d_ws, size_t ws_size,
                              hipStream_t stream) {
    const float* x  = (const float*)d_in[0];
    const float* Wq = (const float*)d_in[1];
    const float* bq = (const float*)d_in[2];
    const float* Wk = (const float*)d_in[3];
    const float* bk = (const float*)d_in[4];
    const float* Wv = (const float*)d_in[5];
    const float* bv = (const float*)d_in[6];
    float* out = (float*)d_out;

    char* ws = (char*)d_ws;
    __bf16* xb  = (__bf16*)(ws);                                  // 16,777,216 B
    __bf16* Wt  = (__bf16*)(ws + 16777216);                       //  1,572,864 B
    __bf16* qbf = (__bf16*)(ws + 18350080);                       //  4,194,304 B
    __bf16* kbf = (__bf16*)(ws + 22544384);                       //  4,194,304 B
    __bf16* vTb = (__bf16*)(ws + 26738688);                       //  4,194,304 B

    pack_kernel<<<2048, 256, 0, stream>>>(x, Wq, Wk, Wv, xb, Wt);
    // (B*T/32) * 3 matrices * (E/64) col groups = 3072 waves / 8 per block
    proj_kernel<<<384, 256, 0, stream>>>(xb, Wt, bq, bk, bv, qbf, kbf, vTb);
    attn_kernel<<<dim3(T_SEQ / 128, BATCH), 256, SMEM_BYTES, stream>>>(qbf, kbf, vTb, out);
}
